// GraphEncoder_41248865911345
// MI455X (gfx1250) — matrix operand order repaired and checked by reading, not compiled, not run on hardware
//
#include <hip/hip_runtime.h>
#include <hip/hip_bf16.h>
#include <stdint.h>

// ---------------------------------------------------------------------------
// GraphSAGE encoder for MI455X (gfx1250, wave32, WMMA).
//
// N=100000 nodes, E=1600000 edges, widths 40 -> 64 -> 128 -> 64 -> 3.
// Edge gather/scatter is the bandwidth bottleneck (~GBs/pass @ 23.3 TB/s);
// activations are kept in bf16 to halve that traffic and feed
// v_wmma_f32_16x16x32_bf16 directly. Scatter-reduction uses hardware
// global_atomic_add_f32 (unsafeAtomicAdd -> coarse-grained workspace).
// ---------------------------------------------------------------------------

typedef __attribute__((ext_vector_type(16))) __bf16 v16bf;
typedef __attribute__((ext_vector_type(8)))  __bf16 v8bf;
typedef __attribute__((ext_vector_type(4)))  __bf16 v4bf;
typedef __attribute__((ext_vector_type(8)))  float  v8f;

// Force the native global_atomic_add_f32 (workspace is coarse-grained device
// memory, so the "unsafe" HW path is exact here; avoids a CAS loop).
__device__ __forceinline__ void atomic_add_f32(float* p, float v) {
#if defined(__HIP_PLATFORM_AMD__)
  unsafeAtomicAdd(p, v);
#else
  atomicAdd(p, v);
#endif
}

// ---------------- degree count ----------------
__global__ void k_count(const int* __restrict__ dst, uint32_t* __restrict__ cnt,
                        long long E) {
  long long i = (long long)blockIdx.x * blockDim.x + threadIdx.x;
  if (i < E) atomicAdd(&cnt[dst[i]], 1u);
}

__global__ void k_inv(const uint32_t* __restrict__ cnt, float* __restrict__ inv,
                      int N) {
  int n = blockIdx.x * blockDim.x + threadIdx.x;
  if (n < N) {
    float c = (float)cnt[n];
    inv[n] = 1.0f / fmaxf(c, 1.0f);
  }
}

// ---------------- input prep: x (N x 40 fp32) -> right half of A1, bf16 padded to 64
__global__ void k_prep_x(const float* __restrict__ x, __bf16* __restrict__ A1,
                         int N) {
  int idx = blockIdx.x * blockDim.x + threadIdx.x;   // over N*64
  if (idx >= N * 64) return;
  int n = idx >> 6, c = idx & 63;
  float v = (c < 40) ? x[n * 40 + c] : 0.0f;
  A1[(size_t)n * 128 + 64 + c] = (__bf16)v;
}

// ---------------- weight prep: Wt[c][k] = stacked/padded [Wl;Wr][k][c] in bf16
__global__ void k_prep_w(const float* __restrict__ Wl, const float* __restrict__ Wr,
                         __bf16* __restrict__ Wt, int d_in, int d_pad, int d_out) {
  int Ktot = 2 * d_pad;
  int idx = blockIdx.x * blockDim.x + threadIdx.x;
  if (idx >= d_out * Ktot) return;
  int c = idx / Ktot, k = idx - c * Ktot;
  float v = 0.0f;
  if (k < d_pad) {
    if (k < d_in) v = Wl[k * d_out + c];
  } else {
    int kk = k - d_pad;
    if (kk < d_in) v = Wr[kk * d_out + c];
  }
  Wt[idx] = (__bf16)v;
}

// ---------------- edge aggregation: agg[dst] += H[src] (4 feats / thread) ----
// H points at the feature block (bf16), row stride S elements, width d=2^?.
__global__ void k_aggregate(const __bf16* __restrict__ H, int S, int d, int cshift,
                            const int* __restrict__ src, const int* __restrict__ dst,
                            float* __restrict__ agg, long long total) {
  long long i = (long long)blockIdx.x * blockDim.x + threadIdx.x;
  if (i >= total) return;
  long long e = i >> cshift;                 // edge id
  int c = (int)(i & ((1 << cshift) - 1)) << 2;   // feature offset (x4)
  int s = src[e], t = dst[e];
  v4bf h = *(const v4bf*)(H + (size_t)s * S + c);
  float* ap = agg + (size_t)t * d + c;
  atomic_add_f32(ap + 0, (float)h[0]);
  atomic_add_f32(ap + 1, (float)h[1]);
  atomic_add_f32(ap + 2, (float)h[2]);
  atomic_add_f32(ap + 3, (float)h[3]);
}

// ---------------- mean = agg * inv  -> left half of A (bf16) ----------------
__global__ void k_mean(const float* __restrict__ agg, const float* __restrict__ inv,
                       __bf16* __restrict__ A, int S, int d, int dshift, int N) {
  int idx = blockIdx.x * blockDim.x + threadIdx.x;    // over N*d
  if (idx >= N * d) return;
  int n = idx >> dshift, c = idx & (d - 1);
  A[(size_t)n * S + c] = (__bf16)(agg[idx] * inv[n]);
}

// ---------------- WMMA GEMM: Out = relu(A @ W + b), bf16 in/out, f32 acc -----
// A: N x Ktot bf16 row-major.  Wt: d_out x Ktot bf16 (pre-transposed weights).
// One wave -> 16x16 tile; block = 4 waves covering 16 rows x 64 cols.
__global__ __launch_bounds__(128) void k_gemm_wmma(
    const __bf16* __restrict__ A, int Ktot,
    const __bf16* __restrict__ Wt,
    const float* __restrict__ bias,
    __bf16* __restrict__ Out, int Sout, int outoff, int relu) {
  const int wave = threadIdx.x >> 5;
  const int lane = threadIdx.x & 31;
  const int lh = lane & 15;          // row/col within tile
  const int hh = lane >> 4;          // half-wave selector
  const int rowbase = blockIdx.x * 16;
  const int colbase = blockIdx.y * 64 + wave * 16;

  const __bf16* __restrict__ arow = A + (size_t)(rowbase + lh) * Ktot;
  const __bf16* __restrict__ wrow = Wt + (size_t)(colbase + lh) * Ktot;

  v8f acc = {};
  for (int k = 0; k < Ktot; k += 32) {
    // A fragment (16x32, MxK): lane<16 holds K 0..7 & 16..23; lane>=16 holds
    // K 8..15 & 24..31 of row M=lane&15 -> two contiguous 16B loads.
    v8bf a0 = *(const v8bf*)(arow + k + hh * 8);
    v8bf a1 = *(const v8bf*)(arow + k + 16 + hh * 8);
    v16bf av = __builtin_shufflevector(a0, a1, 0, 1, 2, 3, 4, 5, 6, 7,
                                       8, 9, 10, 11, 12, 13, 14, 15);
    // B fragment (32x16, KxN): lane holds column N=lane&15; lane<16 holds
    // K 0..7 & 16..23, lane>=16 holds K 8..15 & 24..31 -> two 16B loads from Wt row.
    v8bf b0 = *(const v8bf*)(wrow + k + hh * 8);
    v8bf b1 = *(const v8bf*)(wrow + k + 16 + hh * 8);
    v16bf bv = __builtin_shufflevector(b0, b1, 0, 1, 2, 3, 4, 5, 6, 7,
                                       8, 9, 10, 11, 12, 13, 14, 15);
    acc = __builtin_amdgcn_wmma_f32_16x16x32_bf16(
        /*neg_a=*/false, av, /*neg_b=*/false, bv,
        /*c_mod=*/(short)0, acc, /*reuse_a=*/false, /*reuse_b=*/false);
  }

  // C/D layout: VGPR i -> M = i + 8*hh, N = lane&15.
  const int col = colbase + lh;
  const float bv = bias[col];
#pragma unroll
  for (int i = 0; i < 8; ++i) {
    int m = rowbase + hh * 8 + i;
    float v = acc[i] + bv;
    if (relu) v = fmaxf(v, 0.0f);
    Out[(size_t)m * Sout + outoff + col] = (__bf16)v;
  }
}

// ---------------- final head: out = h3 @ W4 + b4 (64 -> 3, fp32) ------------
__global__ void k_final(const __bf16* __restrict__ H3, const float* __restrict__ W4,
                        const float* __restrict__ b4, float* __restrict__ out,
                        int N) {
  int n = blockIdx.x * blockDim.x + threadIdx.x;
  if (n >= N) return;
  float a0 = b4[0], a1 = b4[1], a2 = b4[2];
  const __bf16* h = H3 + (size_t)n * 64;
#pragma unroll 8
  for (int k = 0; k < 64; ++k) {
    float hv = (float)h[k];
    a0 += hv * W4[k * 3 + 0];
    a1 += hv * W4[k * 3 + 1];
    a2 += hv * W4[k * 3 + 2];
  }
  out[n * 3 + 0] = a0;
  out[n * 3 + 1] = a1;
  out[n * 3 + 2] = a2;
}

// ---------------------------------------------------------------------------
extern "C" void kernel_launch(void* const* d_in, const int* in_sizes, int n_in,
                              void* d_out, int out_size, void* d_ws, size_t ws_size,
                              hipStream_t stream) {
  const int N = in_sizes[0] / 40;
  const long long E = in_sizes[1] / 2;

  const float* x   = (const float*)d_in[0];
  const int*   ei  = (const int*)d_in[1];
  const float* W1l = (const float*)d_in[2];
  const float* b1  = (const float*)d_in[3];
  const float* W1r = (const float*)d_in[4];
  const float* W2l = (const float*)d_in[5];
  const float* b2  = (const float*)d_in[6];
  const float* W2r = (const float*)d_in[7];
  const float* W3l = (const float*)d_in[8];
  const float* b3  = (const float*)d_in[9];
  const float* W3r = (const float*)d_in[10];
  const float* W4  = (const float*)d_in[11];
  const float* b4  = (const float*)d_in[12];
  const int* src = ei;
  const int* dst = ei + E;
  float* out = (float*)d_out;

  // Workspace carve-up (~167 MB for N=100000).
  __bf16* A1 = (__bf16*)d_ws;               // N x 128: [mean(x)|x_pad]
  __bf16* A2 = A1 + (size_t)N * 128;        // N x 128: [mean(h1)|h1]
  __bf16* A3 = A2 + (size_t)N * 128;        // N x 256: [mean(h2)|h2]
  __bf16* H3 = A3 + (size_t)N * 256;        // N x 64
  float* agg = (float*)(H3 + (size_t)N * 64);  // N x 128 fp32
  float* inv = agg + (size_t)N * 128;          // N
  uint32_t* cnt = (uint32_t*)(inv + N);        // N
  __bf16* Wt1 = (__bf16*)(cnt + N);            // 64 x 128
  __bf16* Wt2 = Wt1 + 64 * 128;                // 128 x 128
  __bf16* Wt3 = Wt2 + 128 * 128;               // 64 x 256

  const int TB = 256;

  // Degrees (shared by all layers).
  hipMemsetAsync(cnt, 0, (size_t)N * 4, stream);
  k_count<<<(int)((E + TB - 1) / TB), TB, 0, stream>>>(dst, cnt, E);
  k_inv<<<(N + TB - 1) / TB, TB, 0, stream>>>(cnt, inv, N);

  // Input + weight prep.
  k_prep_x<<<(N * 64 + TB - 1) / TB, TB, 0, stream>>>(x, A1, N);
  k_prep_w<<<(64 * 128 + TB - 1) / TB, TB, 0, stream>>>(W1l, W1r, Wt1, 40, 64, 64);
  k_prep_w<<<(128 * 128 + TB - 1) / TB, TB, 0, stream>>>(W2l, W2r, Wt2, 64, 64, 128);
  k_prep_w<<<(64 * 256 + TB - 1) / TB, TB, 0, stream>>>(W3l, W3r, Wt3, 128, 128, 64);

  // ---- Layer 1: d=64 (padded), Ktot=128, out 64 -> right half of A2 ----
  {
    hipMemsetAsync(agg, 0, (size_t)N * 64 * 4, stream);
    long long tot = E * 16;  // d/4 = 16 chunks
    k_aggregate<<<(int)((tot + TB - 1) / TB), TB, 0, stream>>>(
        A1 + 64, 128, 64, 4, src, dst, agg, tot);
    k_mean<<<(N * 64 + TB - 1) / TB, TB, 0, stream>>>(agg, inv, A1, 128, 64, 6, N);
    dim3 g(N / 16, 64 / 64);
    k_gemm_wmma<<<g, 128, 0, stream>>>(A1, 128, Wt1, b1, A2, 128, 64, 1);
  }

  // ---- Layer 2: d=64, Ktot=128, out 128 -> right half of A3 ----
  {
    hipMemsetAsync(agg, 0, (size_t)N * 64 * 4, stream);
    long long tot = E * 16;
    k_aggregate<<<(int)((tot + TB - 1) / TB), TB, 0, stream>>>(
        A2 + 64, 128, 64, 4, src, dst, agg, tot);
    k_mean<<<(N * 64 + TB - 1) / TB, TB, 0, stream>>>(agg, inv, A2, 128, 64, 6, N);
    dim3 g(N / 16, 128 / 64);
    k_gemm_wmma<<<g, 128, 0, stream>>>(A2, 128, Wt2, b2, A3, 256, 128, 1);
  }

  // ---- Layer 3: d=128, Ktot=256, out 64 -> H3 ----
  {
    hipMemsetAsync(agg, 0, (size_t)N * 128 * 4, stream);
    long long tot = E * 32;  // d/4 = 32 chunks
    k_aggregate<<<(int)((tot + TB - 1) / TB), TB, 0, stream>>>(
        A3 + 128, 256, 128, 5, src, dst, agg, tot);
    k_mean<<<(N * 128 + TB - 1) / TB, TB, 0, stream>>>(agg, inv, A3, 256, 128, 7, N);
    dim3 g(N / 16, 64 / 64);
    k_gemm_wmma<<<g, 128, 0, stream>>>(A3, 256, Wt3, b3, H3, 64, 0, 1);
  }

  // ---- Final head ----
  k_final<<<(N + TB - 1) / TB, TB, 0, stream>>>(H3, W4, b4, out, N);
}
